// AbstractGenerator_75445395522265
// MI455X (gfx1250) — compile-verified, matrix operand order checked by reading, not run on hardware
//
#include <hip/hip_runtime.h>

// Problem constants (fixed by the reference): B=64, S=128, D=1024, V=32000, max_len=50
#define BB 64
#define SS 128
#define DD 1024
#define VV 32000
#define TT 50

typedef __attribute__((ext_vector_type(16))) __bf16    v16bf;
typedef __attribute__((ext_vector_type(8)))  __bf16    v8bf;
typedef __attribute__((ext_vector_type(8)))  float     v8f;
typedef __attribute__((ext_vector_type(8)))  unsigned  v8u;

// RNE fp32 -> bf16 (used outside inner GEMM loops only)
__device__ __forceinline__ __bf16 f2bf(float f) {
    unsigned u = __float_as_uint(f);
    unsigned r = u + 0x7FFFu + ((u >> 16) & 1u);
    unsigned short hs = (unsigned short)(r >> 16);
    return __builtin_bit_cast(__bf16, hs);
}

// ---- 16x32 (MxK) bf16 fragment loaders -----------------------------------
// CDNA5 16-bit A/B layout: lanes 0-15 hold K = kk+[0..7] and kk+[16..23],
// lanes 16-31 hold K = kk+[8..15] and kk+[24..31]. rowPtr is this lane's row.

// From pre-converted bf16 rows: two b128 loads, zero VALU.
__device__ __forceinline__ v16bf load_frag_bf(const __bf16* __restrict__ rowPtr, int kk, int g) {
    int kb = kk + (g << 3);
    v8bf lo = *(const v8bf*)(rowPtr + kb);
    v8bf hi = *(const v8bf*)(rowPtr + kb + 16);
    return __builtin_shufflevector(lo, hi, 0, 1, 2, 3, 4, 5, 6, 7,
                                   8, 9, 10, 11, 12, 13, 14, 15);
}

// From fp32 rows: four b128 loads + 8x v_perm_b32 (truncate-to-bf16 packing).
__device__ __forceinline__ v16bf load_frag_f32(const float* __restrict__ rowPtr, int kk, int g) {
    int kb = kk + (g << 3);
    const uint4* p = (const uint4*)(rowPtr + kb);
    uint4 x0 = p[0], x1 = p[1];
    const uint4* q = (const uint4*)(rowPtr + kb + 16);
    uint4 x2 = q[0], x3 = q[1];
    v8u r;
    r[0] = __builtin_amdgcn_perm(x0.y, x0.x, 0x07060302u);
    r[1] = __builtin_amdgcn_perm(x0.w, x0.z, 0x07060302u);
    r[2] = __builtin_amdgcn_perm(x1.y, x1.x, 0x07060302u);
    r[3] = __builtin_amdgcn_perm(x1.w, x1.z, 0x07060302u);
    r[4] = __builtin_amdgcn_perm(x2.y, x2.x, 0x07060302u);
    r[5] = __builtin_amdgcn_perm(x2.w, x2.z, 0x07060302u);
    r[6] = __builtin_amdgcn_perm(x3.y, x3.x, 0x07060302u);
    r[7] = __builtin_amdgcn_perm(x3.w, x3.z, 0x07060302u);
    return __builtin_bit_cast(v16bf, r);
}

__device__ __forceinline__ v8f wmma_bf16(v16bf a, v16bf b, v8f c) {
    return __builtin_amdgcn_wmma_f32_16x16x32_bf16(false, a, false, b, (short)0, c, false, false);
}

__device__ __forceinline__ float sigm(float x) { return 1.0f / (1.0f + __expf(-x)); }

// ---------------------------------------------------------------------------
// one-time: fp32 -> bf16 weight conversion (grid-stride)
// ---------------------------------------------------------------------------
__global__ void cvt_kernel(const float* __restrict__ src, __bf16* __restrict__ dst, int n) {
    int stride = gridDim.x * blockDim.x;
    for (int i = blockIdx.x * blockDim.x + threadIdx.x; i < n; i += stride)
        dst[i] = f2bf(src[i]);
}

// ---------------------------------------------------------------------------
// K0: zero h/c/h_bf/tok; sel_term[b] = mean_s(selected[b,s,:]) . Wc[0,D:] + bc
// ---------------------------------------------------------------------------
__global__ void init_kernel(const float* __restrict__ selected,
                            const float* __restrict__ Wc,
                            const float* __restrict__ bc,
                            float* __restrict__ h, float* __restrict__ c,
                            __bf16* __restrict__ hBf,
                            int* __restrict__ tok, float* __restrict__ selTerm) {
    int b = blockIdx.x, tid = threadIdx.x;
    __shared__ float red[256];
    for (int d = tid; d < DD; d += 256) {
        h[b * DD + d] = 0.0f; c[b * DD + d] = 0.0f;
        hBf[b * DD + d] = __builtin_bit_cast(__bf16, (unsigned short)0);
    }
    float partial = 0.0f;
    for (int i = tid; i < SS * DD; i += 256) {
        int d = i & (DD - 1);
        partial += selected[(size_t)b * SS * DD + i] * Wc[DD + d];
    }
    red[tid] = partial; __syncthreads();
    for (int s = 128; s > 0; s >>= 1) { if (tid < s) red[tid] += red[tid + s]; __syncthreads(); }
    if (tid == 0) { selTerm[b] = red[0] * (1.0f / (float)SS) + bc[0]; tok[b] = 0; }
}

// ---------------------------------------------------------------------------
// per step: gather emb_bf[b,:] = bf16(emb_W[tok[b],:])
// ---------------------------------------------------------------------------
__global__ void gather_kernel(const float* __restrict__ embW,
                              const int* __restrict__ tok,
                              __bf16* __restrict__ embBf) {
    int b = blockIdx.x;
    size_t base = (size_t)tok[b] * DD;
    for (int d = threadIdx.x; d < DD; d += 256)
        embBf[b * DD + d] = f2bf(embW[base + d]);
}

// ---------------------------------------------------------------------------
// K1: gates[64,4096] = emb @ W_ih^T + h @ W_hh^T + (b_ih+b_hh)
// split-K x4: 1024 tiles x 4 chunks = 4096 waves -> 512 blocks x 256 threads
// ---------------------------------------------------------------------------
template <bool PRE>
__global__ void gates_kernel(const __bf16* __restrict__ embBf,
                             const __bf16* __restrict__ hBf,
                             const __bf16* __restrict__ WihBf,
                             const __bf16* __restrict__ WhhBf,
                             const float* __restrict__ Wih,
                             const float* __restrict__ Whh,
                             const float* __restrict__ bih,
                             const float* __restrict__ bhh,
                             float* __restrict__ gates) {
    __shared__ float lds[8 * 256];
    int wave = threadIdx.x >> 5, lane = threadIdx.x & 31;
    int tileLocal = wave >> 2, chunk = wave & 3;
    int tileIdx = blockIdx.x * 2 + tileLocal;     // 0..1023
    int mt = tileIdx & 3, nt = tileIdx >> 2;      // 4 m-tiles x 256 n-tiles
    int g = lane >> 4, ln = lane & 15;

    int brow = mt * 16 + ln;
    const __bf16* ePtr = embBf + brow * DD;
    const __bf16* hPtr = hBf + brow * DD;
    int nrow = nt * 16 + ln;
    const __bf16* wiB = WihBf + (size_t)nrow * DD;
    const __bf16* whB = WhhBf + (size_t)nrow * DD;
    const float*  wiF = Wih + (size_t)nrow * DD;
    const float*  whF = Whh + (size_t)nrow * DD;

    v8f acc = {};
    int k0 = chunk * (DD / 4);
    for (int kk = k0; kk < k0 + DD / 4; kk += 32) {
        v16bf ae = load_frag_bf(ePtr, kk, g);
        v16bf bi = PRE ? load_frag_bf(wiB, kk, g) : load_frag_f32(wiF, kk, g);
        acc = wmma_bf16(ae, bi, acc);
        v16bf ah = load_frag_bf(hPtr, kk, g);
        v16bf bh = PRE ? load_frag_bf(whB, kk, g) : load_frag_f32(whF, kk, g);
        acc = wmma_bf16(ah, bh, acc);
    }
    if (chunk != 0) {
#pragma unroll
        for (int r = 0; r < 8; ++r) lds[wave * 256 + lane * 8 + r] = acc[r];
    }
    __syncthreads();
    if (chunk == 0) {
#pragma unroll
        for (int r = 0; r < 8; ++r)
            acc[r] += lds[(wave + 1) * 256 + lane * 8 + r] +
                      lds[(wave + 2) * 256 + lane * 8 + r] +
                      lds[(wave + 3) * 256 + lane * 8 + r];
        float bias = bih[nrow] + bhh[nrow];
#pragma unroll
        for (int r = 0; r < 8; ++r) {
            int m = mt * 16 + r + (g << 3);       // C/D layout: m = r + 8*(lane/16)
            gates[(size_t)m * (4 * DD) + nrow] = acc[r] + bias;
        }
    }
}

// ---------------------------------------------------------------------------
// K2: LSTM cell + copy-score dot h.Wc_h ; writes h, h_bf ; resets argmax keys
// ---------------------------------------------------------------------------
__global__ void cell_kernel(const float* __restrict__ gates,
                            const float* __restrict__ Wc,
                            const float* __restrict__ selTerm,
                            float* __restrict__ h, float* __restrict__ c,
                            __bf16* __restrict__ hBf,
                            float* __restrict__ copyBase,
                            unsigned long long* __restrict__ amax) {
    int b = blockIdx.x, tid = threadIdx.x;
    __shared__ float red[256];
    const float* gr = gates + (size_t)b * (4 * DD);
    float partial = 0.0f;
    for (int d = tid; d < DD; d += 256) {
        float gi = gr[d], gf = gr[DD + d], gg = gr[2 * DD + d], go = gr[3 * DD + d];
        float cn = sigm(gf) * c[b * DD + d] + sigm(gi) * tanhf(gg);
        float hn = sigm(go) * tanhf(cn);
        c[b * DD + d] = cn;
        h[b * DD + d] = hn;
        hBf[b * DD + d] = f2bf(hn);
        partial += hn * Wc[d];
    }
    red[tid] = partial; __syncthreads();
    for (int s = 128; s > 0; s >>= 1) { if (tid < s) red[tid] += red[tid + s]; __syncthreads(); }
    if (tid == 0) { copyBase[b] = red[0] + selTerm[b]; amax[b] = 0ull; }
}

// ---------------------------------------------------------------------------
// K3: logits = h @ Wo^T + bo + copyBase ; store out[b,step,:]; packed argmax
// split-K x2: 8000 tiles x 2 chunks = 16000 waves -> 2000 blocks x 256 threads
// ---------------------------------------------------------------------------
template <bool PRE>
__global__ void logits_kernel(const __bf16* __restrict__ WoBf,
                              const float* __restrict__ Wo,
                              const __bf16* __restrict__ hBf,
                              const float* __restrict__ bo,
                              const float* __restrict__ copyBase,
                              float* __restrict__ out,
                              unsigned long long* __restrict__ amax,
                              int step) {
    __shared__ float lds[8 * 256];
    int wave = threadIdx.x >> 5, lane = threadIdx.x & 31;
    int tileLocal = wave >> 1, chunk = wave & 1;
    int tileIdx = blockIdx.x * 4 + tileLocal;     // 0..7999
    int mt = tileIdx & 3, nt = tileIdx >> 2;      // 4 m-tiles x 2000 n-tiles
    int g = lane >> 4, ln = lane & 15;

    int brow = mt * 16 + ln;
    const __bf16* hPtr = hBf + brow * DD;
    int vrow = nt * 16 + ln;
    const __bf16* wB = WoBf + (size_t)vrow * DD;
    const float*  wF = Wo + (size_t)vrow * DD;

    v8f acc = {};
    int k0 = chunk * (DD / 2);
    for (int kk = k0; kk < k0 + DD / 2; kk += 32) {
        v16bf a = load_frag_bf(hPtr, kk, g);
        v16bf b = PRE ? load_frag_bf(wB, kk, g) : load_frag_f32(wF, kk, g);
        acc = wmma_bf16(a, b, acc);
    }
    if (chunk != 0) {
#pragma unroll
        for (int r = 0; r < 8; ++r) lds[wave * 256 + lane * 8 + r] = acc[r];
    }
    __syncthreads();
    if (chunk == 0) {
#pragma unroll
        for (int r = 0; r < 8; ++r) acc[r] += lds[(wave + 1) * 256 + lane * 8 + r];
        float bov = bo[vrow];
#pragma unroll
        for (int r = 0; r < 8; ++r) {
            int m = mt * 16 + r + (g << 3);       // batch index
            float val = acc[r] + bov + copyBase[m];
            out[((size_t)m * TT + step) * VV + vrow] = val;

            // per-row max over this half-wave's 16 columns
            float mv = val; int mi = vrow;
            for (int off = 8; off > 0; off >>= 1) {
                float ov = __shfl_xor(mv, off, 16);
                int   oi = __shfl_xor(mi, off, 16);
                if (ov > mv || (ov == mv && oi < mi)) { mv = ov; mi = oi; }
            }
            if (ln == 0) {
                unsigned fu = __float_as_uint(mv);
                fu = (fu & 0x80000000u) ? ~fu : (fu | 0x80000000u);  // order-preserving
                unsigned long long key = ((unsigned long long)fu << 32) |
                                         (unsigned long long)(0xFFFFFFFFu - (unsigned)mi);
                atomicMax(amax + m, key);
            }
        }
    }
}

// K4: decode winning token index
__global__ void tok_kernel(const unsigned long long* __restrict__ amax, int* __restrict__ tok) {
    int b = threadIdx.x;
    if (b < BB) tok[b] = (int)(0xFFFFFFFFu - (unsigned)(amax[b] & 0xFFFFFFFFull));
}

// ---------------------------------------------------------------------------
// workspace layout (bytes)
// ---------------------------------------------------------------------------
#define WS_H        0u          // 64*1024 f32    = 262144
#define WS_C        262144u     // 64*1024 f32
#define WS_GATES    524288u     // 64*4096 f32    = 1048576
#define WS_SELTERM  1572864u    // 64 f32
#define WS_COPYB    1573120u    // 64 f32
#define WS_TOK      1573376u    // 64 i32
#define WS_AMAX     1573632u    // 64 u64
#define WS_HBF      1574144u    // 64*1024 bf16   = 131072
#define WS_EMBBF    1705216u    // 64*1024 bf16
#define WS_WIHBF    1836288u    // 4096*1024 bf16 = 8388608
#define WS_WHHBF    10224896u   // 4096*1024 bf16
#define WS_WOBF     18613504u   // 32000*1024 bf16 = 65536000
#define WS_PRE_REQ  84149504ull

extern "C" void kernel_launch(void* const* d_in, const int* in_sizes, int n_in,
                              void* d_out, int out_size, void* d_ws, size_t ws_size,
                              hipStream_t stream) {
    const float* selected = (const float*)d_in[0];   // (B,S,D)
    const float* embW     = (const float*)d_in[1];   // (V,D)
    const float* Wih      = (const float*)d_in[2];   // (4D,D)
    const float* Whh      = (const float*)d_in[3];   // (4D,D)
    const float* bih      = (const float*)d_in[4];   // (4D,)
    const float* bhh      = (const float*)d_in[5];   // (4D,)
    const float* Wc       = (const float*)d_in[6];   // (1,2D)
    const float* bc       = (const float*)d_in[7];   // (1,)
    const float* Wo       = (const float*)d_in[8];   // (V,D)
    const float* bo       = (const float*)d_in[9];   // (V,)
    (void)in_sizes; (void)n_in; (void)out_size;      // d_in[10]=max_len, fixed at 50

    char* ws = (char*)d_ws;
    float* h        = (float*)(ws + WS_H);
    float* c        = (float*)(ws + WS_C);
    float* gates    = (float*)(ws + WS_GATES);
    float* selTerm  = (float*)(ws + WS_SELTERM);
    float* copyBase = (float*)(ws + WS_COPYB);
    int*   tok      = (int*)  (ws + WS_TOK);
    unsigned long long* amax = (unsigned long long*)(ws + WS_AMAX);
    __bf16* hBf     = (__bf16*)(ws + WS_HBF);
    __bf16* embBf   = (__bf16*)(ws + WS_EMBBF);
    __bf16* WihBf   = (__bf16*)(ws + WS_WIHBF);
    __bf16* WhhBf   = (__bf16*)(ws + WS_WHHBF);
    __bf16* WoBf    = (__bf16*)(ws + WS_WOBF);
    float* out = (float*)d_out;

    const bool pre = (ws_size >= WS_PRE_REQ);
    if (pre) {
        cvt_kernel<<<1024, 256, 0, stream>>>(Wih, WihBf, 4 * DD * DD);
        cvt_kernel<<<1024, 256, 0, stream>>>(Whh, WhhBf, 4 * DD * DD);
        cvt_kernel<<<4096, 256, 0, stream>>>(Wo, WoBf, VV * DD);
    }
    init_kernel<<<BB, 256, 0, stream>>>(selected, Wc, bc, h, c, hBf, tok, selTerm);

    for (int t = 0; t < TT; ++t) {
        gather_kernel<<<BB, 256, 0, stream>>>(embW, tok, embBf);
        if (pre)
            gates_kernel<true><<<512, 256, 0, stream>>>(embBf, hBf, WihBf, WhhBf,
                                                        Wih, Whh, bih, bhh, gates);
        else
            gates_kernel<false><<<512, 256, 0, stream>>>(embBf, hBf, WihBf, WhhBf,
                                                         Wih, Whh, bih, bhh, gates);
        cell_kernel<<<BB, 256, 0, stream>>>(gates, Wc, selTerm, h, c, hBf, copyBase, amax);
        if (pre)
            logits_kernel<true><<<2000, 256, 0, stream>>>(WoBf, Wo, hBf, bo, copyBase,
                                                          out, amax, t);
        else
            logits_kernel<false><<<2000, 256, 0, stream>>>(WoBf, Wo, hBf, bo, copyBase,
                                                           out, amax, t);
        tok_kernel<<<1, 64, 0, stream>>>(amax, tok);
    }
}
